// Qwen3VLSequentialMoeTextExperts_38474317038321
// MI455X (gfx1250) — matrix operand order repaired and checked by reading, not compiled, not run on hardware
//
#include <hip/hip_runtime.h>
#include <math.h>

#define E_  32
#define H_  2048
#define F_  768
#define F2_ 1536
#define T_  2048
#define HK  1024          // K chunk of x staged in LDS at a time
#define XLD (HK + 8)      // xh row stride (halves) -> de-conflict LDS banks
#define ALD (F_ + 8)      // act row stride (halves)

typedef __attribute__((ext_vector_type(4)))  _Float16 v4h;
typedef __attribute__((ext_vector_type(8)))  _Float16 v8h;
typedef __attribute__((ext_vector_type(16))) _Float16 v16h;
typedef __attribute__((ext_vector_type(8)))  float    v8f;

__device__ __forceinline__ v8f wmma_f16(v16h a, v16h b, v8f c) {
  // 8 args: (neg_a, A, neg_b, B, c_mod, C, reuse_a, reuse_b)
  return __builtin_amdgcn_wmma_f32_16x16x32_f16(false, a, false, b, (short)0, c,
                                                false, false);
}

__device__ __forceinline__ v16h cat8(v8h lo, v8h hi) {
  return __builtin_shufflevector(lo, hi, 0, 1, 2, 3, 4, 5, 6, 7,
                                 8, 9, 10, 11, 12, 13, 14, 15);
}

// Build a 32x16 f16 B-fragment from an fp32 row-major matrix (row stride ld).
// Caller passes p already offset to element (k_base_of_this_lane, n_of_this_lane).
// Contiguous k order used here (A matches): lanes 0-15: N=lane, K=0..15 in halves; lanes 16-31: N=lane-16, K=16..31.
__device__ __forceinline__ v16h loadB16_f32(const float* __restrict__ p, int ld) {
  v16h b;
#pragma unroll
  for (int h = 0; h < 16; ++h) b[h] = (_Float16)p[(size_t)h * ld];
  return b;
}

// ---------------- weight repack: fp32 (k-major) -> f16 fragment-order -------
// W1 frags: [E][96 n-tiles][64 k-steps][512 halves]; W3: [E][128][24][512]
__global__ __launch_bounds__(256) void repack_w1(const float* __restrict__ w1,
                                                 _Float16* __restrict__ w1p) {
  const int frag = blockIdx.x * 8 + (threadIdx.x >> 5);
  const int lane = threadIdx.x & 31;
  const int ks = frag & 63;
  const int j  = (frag >> 6) % 96;
  const int e  = frag / (64 * 96);
  const int n  = j * 16 + (lane & 15);
  const int k0 = ks * 32 + ((lane >> 4) << 4);
  v16h b = loadB16_f32(w1 + ((size_t)e * H_ + k0) * F2_ + n, F2_);
  *(v16h*)(w1p + ((size_t)frag << 9) + lane * 16) = b;
}

__global__ __launch_bounds__(256) void repack_w3(const float* __restrict__ w3,
                                                 _Float16* __restrict__ w3p) {
  const int frag = blockIdx.x * 8 + (threadIdx.x >> 5);
  const int lane = threadIdx.x & 31;
  const int ks = frag % 24;
  const int j  = (frag / 24) % 128;
  const int e  = frag / (24 * 128);
  const int n  = j * 16 + (lane & 15);
  const int k0 = ks * 32 + ((lane >> 4) << 4);
  v16h b = loadB16_f32(w3 + ((size_t)e * F_ + k0) * H_ + n, H_);
  *(v16h*)(w3p + ((size_t)frag << 9) + lane * 16) = b;
}

// ---------------- fused MoE kernel ------------------------------------------
// block = 256 threads (8 waves); blockIdx.x = token tile (16 tokens), .y = expert
template <bool PACKED>
__global__ __launch_bounds__(256)
void moe_fused(const float* __restrict__ x, const float* __restrict__ rw,
               const float* __restrict__ w1, const float* __restrict__ w3,
               const _Float16* __restrict__ w1p, const _Float16* __restrict__ w3p,
               float* __restrict__ out) {
  __shared__ _Float16 xh[16 * XLD];    // 33,024 B
  __shared__ _Float16 acth[16 * ALD];  // 24,832 B   (total 57,856 B LDS)

  const int tid  = threadIdx.x;
  const int wave = tid >> 5;
  const int lane = tid & 31;
  const int e    = blockIdx.y;
  const int t0   = blockIdx.x * 16;

  const int m     = lane & 15;   // N column (B/C/D) or M row (A) for this lane
  const int hi16  = lane >> 4;   // lane half
  const int kbase = hi16 * 8;    // A-fragment K offset per ISA layout
  const int kb16  = hi16 * 16;   // A/B-fragment K offset (contiguous k order)

  // ---------------- stage 1: gate_up = x_tile @ W1[e] ----------------------
  v8f zero = {};
  v8f accg[6], accu[6];
#pragma unroll
  for (int i = 0; i < 6; ++i) { accg[i] = zero; accu[i] = zero; }

  const float* w1e = w1 + (size_t)e * H_ * F2_;

  for (int khalf = 0; khalf < 2; ++khalf) {
    __syncthreads();  // previous chunk's reads finished before overwrite
    // cooperative f32->f16 load of x tile columns [khalf*HK, khalf*HK+HK)
    for (int i = tid; i < 16 * HK / 4; i += 256) {
      const int r = i >> 8;              // HK/4 == 256 float4 per row
      const int c = (i & 255) << 2;
      const float4 v =
          *(const float4*)(x + (size_t)(t0 + r) * H_ + khalf * HK + c);
      v4h hv = {(_Float16)v.x, (_Float16)v.y, (_Float16)v.z, (_Float16)v.w};
      *(v4h*)&xh[r * XLD + c] = hv;
    }
    __syncthreads();

    for (int kk = 0; kk < HK; kk += 32) {
      // A fragment (16x32 f16) from LDS: 2 x ds_load_b128 per lane
      v8h lo = *(const v8h*)&xh[m * XLD + kk + kb16];
      v8h hi = *(const v8h*)&xh[m * XLD + kk + kb16 + 8];
      v16h a = cat8(lo, hi);
      const int kg = khalf * HK + kk;
      const int ks = kg >> 5;
#pragma unroll
      for (int i = 0; i < 6; ++i) {
        const int jg = wave * 6 + i;       // gate n-tile  (f = 16*jg + m)
        const int ju = 48 + wave * 6 + i;  // matching up n-tile (f + 768)
        v16h bg, bu;
        if (PACKED) {
          bg = *(const v16h*)(w1p + ((((size_t)e * 96 + jg) * 64 + ks) << 9) +
                              lane * 16);
          bu = *(const v16h*)(w1p + ((((size_t)e * 96 + ju) * 64 + ks) << 9) +
                              lane * 16);
        } else {
          const float* p = w1e + (size_t)(kg + kb16) * F2_;
          bg = loadB16_f32(p + jg * 16 + m, F2_);
          bu = loadB16_f32(p + ju * 16 + m, F2_);
        }
        accg[i] = wmma_f16(a, bg, accg[i]);
        accu[i] = wmma_f16(a, bu, accu[i]);
      }
    }
  }

  // ---------------- silu(gate)*up -> f16 act tile in LDS --------------------
  // C layout: VGPR r holds row M = r + 8*hi16, column N = m
#pragma unroll
  for (int i = 0; i < 6; ++i) {
#pragma unroll
    for (int r = 0; r < 8; ++r) {
      const float g = accg[i][r];
      const float u = accu[i][r];
      const float s = g / (1.0f + __expf(-g));
      acth[(r + 8 * hi16) * ALD + (wave * 6 + i) * 16 + m] = (_Float16)(s * u);
    }
  }
  __syncthreads();

  // ---------------- stage 2: expert_out = act @ W3[e] -----------------------
  v8f acc2[16];
#pragma unroll
  for (int i = 0; i < 16; ++i) acc2[i] = zero;
  const float* w3e = w3 + (size_t)e * F_ * H_;

  for (int fk = 0; fk < F_; fk += 32) {
    v8h lo = *(const v8h*)&acth[m * ALD + fk + kb16];
    v8h hi = *(const v8h*)&acth[m * ALD + fk + kb16 + 8];
    v16h a = cat8(lo, hi);
    const int ks = fk >> 5;
#pragma unroll
    for (int i = 0; i < 16; ++i) {
      const int j2 = wave * 16 + i;  // n-tile over H
      v16h b;
      if (PACKED) {
        b = *(const v16h*)(w3p + ((((size_t)e * 128 + j2) * 24 + ks) << 9) +
                           lane * 16);
      } else {
        b = loadB16_f32(w3e + (size_t)(fk + kb16) * H_ + j2 * 16 + m, H_);
      }
      acc2[i] = wmma_f16(a, b, acc2[i]);
    }
  }

  // ---------------- weighted combine across experts (L2 fp32 atomics) -------
  float wr[8];
#pragma unroll
  for (int r = 0; r < 8; ++r)
    wr[r] = rw[(size_t)(t0 + r + 8 * hi16) * E_ + e];

#pragma unroll
  for (int i = 0; i < 16; ++i) {
    const int n = (wave * 16 + i) * 16 + m;
#pragma unroll
    for (int r = 0; r < 8; ++r) {
      float* dst = out + (size_t)(t0 + r + 8 * hi16) * H_ + n;
      __hip_atomic_fetch_add(dst, acc2[i][r] * wr[r], __ATOMIC_RELAXED,
                             __HIP_MEMORY_SCOPE_AGENT);
    }
  }
}

// ---------------------------------------------------------------------------
extern "C" void kernel_launch(void* const* d_in, const int* in_sizes, int n_in,
                              void* d_out, int out_size, void* d_ws,
                              size_t ws_size, hipStream_t stream) {
  const float* x  = (const float*)d_in[0];  // hidden_states (B,S,H) == (T,H)
  const float* rw = (const float*)d_in[1];  // routing_weights (T,E)
  // d_in[2] router_indices: unused — reference is dense over all experts
  const float* w1 = (const float*)d_in[3];  // gate_up_proj (E,H,2F)
  const float* w3 = (const float*)d_in[4];  // down_proj   (E,F,H)
  float* out = (float*)d_out;

  hipMemsetAsync(out, 0, sizeof(float) * (size_t)T_ * H_, stream);

  const size_t w1p_bytes = (size_t)E_ * 96 * 64 * 512 * sizeof(_Float16);
  const size_t w3p_bytes = (size_t)E_ * 128 * 24 * 512 * sizeof(_Float16);
  const bool packed = (d_ws != nullptr) && (ws_size >= w1p_bytes + w3p_bytes);

  dim3 grid(T_ / 16, E_);
  dim3 block(256);

  if (packed) {
    _Float16* w1p = (_Float16*)d_ws;
    _Float16* w3p = (_Float16*)((char*)d_ws + w1p_bytes);
    repack_w1<<<(E_ * 96 * 64) / 8, 256, 0, stream>>>(w1, w1p);
    repack_w3<<<(E_ * 128 * 24) / 8, 256, 0, stream>>>(w3, w3p);
    moe_fused<true><<<grid, block, 0, stream>>>(x, rw, w1, w3, w1p, w3p, out);
  } else {
    moe_fused<false><<<grid, block, 0, stream>>>(x, rw, w1, w3, nullptr,
                                                 nullptr, out);
  }
}
